// LePEAttention_54889682043088
// MI455X (gfx1250) — matrix operand order repaired and checked by reading, not compiled, not run on hardware
//
#include <hip/hip_runtime.h>

// ---------------- problem constants ----------------
#define RES_    56
#define BATCH_  32
#define CH_     256
#define HEADS_  8
#define HD_     32
#define WSP_    7
#define HSP_    56
#define S_TOK   392        // HSP*WSP tokens per window
#define S_PAD   416        // 13 * 32
#define NW_     8          // RES/WSP windows per image
#define NJOBS   13         // ceil(416/32) 32-row query jobs
#define KCHUNKS 13         // S_PAD / 32 key chunks

typedef __attribute__((ext_vector_type(16))) __bf16 v16bf;
typedef __attribute__((ext_vector_type(8)))  float  v8f;

union FragB16 {
    v16bf v;
    uint4 q[2];
    unsigned short us[16];
};

__device__ __forceinline__ unsigned short f2bf(float f) {
    unsigned u = __builtin_bit_cast(unsigned, f);
    return (unsigned short)((u + 0x7FFFu + ((u >> 16) & 1u)) >> 16);
}
__device__ __forceinline__ float bf2f(unsigned short h) {
    unsigned u = ((unsigned)h) << 16;
    return __builtin_bit_cast(float, u);
}

// DPP row_ror rotation within 16-lane rows (VALU, no LDS traffic)
template <int CTRL>
__device__ __forceinline__ float dpp_rotf(float x) {
    int r = __builtin_amdgcn_update_dpp(0, __builtin_bit_cast(int, x),
                                        CTRL, 0xF, 0xF, true);
    return __builtin_bit_cast(float, r);
}
__device__ __forceinline__ float rowmax16(float x) {
    x = fmaxf(x, dpp_rotf<0x121>(x));   // row_ror:1
    x = fmaxf(x, dpp_rotf<0x122>(x));   // row_ror:2
    x = fmaxf(x, dpp_rotf<0x124>(x));   // row_ror:4
    x = fmaxf(x, dpp_rotf<0x128>(x));   // row_ror:8
    return x;
}
__device__ __forceinline__ float rowsum16(float x) {
    x += dpp_rotf<0x121>(x);
    x += dpp_rotf<0x122>(x);
    x += dpp_rotf<0x124>(x);
    x += dpp_rotf<0x128>(x);
    return x;
}

__global__ __launch_bounds__(256)
void lepe_attn_kernel(const float* __restrict__ q,
                      const float* __restrict__ k,
                      const float* __restrict__ v,
                      const float* __restrict__ wconv,   // (3,3,1,C) HWIO
                      const float* __restrict__ bconv,   // (C,)
                      float* __restrict__ out)           // (B, N, C)
{
    __shared__ unsigned short Kb[S_PAD * HD_];       // [key][d]   bf16
    __shared__ unsigned short Vt[HD_ * S_PAD];       // [d][key]   bf16 (transposed)
    __shared__ unsigned short Pbuf[8][2][16 * 32];   // per-wave P scratch, 2 tiles

    const int bw   = blockIdx.x;          // 0..255  window index (b*8 + wx)
    const int h    = blockIdx.y;          // 0..7    head
    const int b    = bw >> 3;
    const int wx   = bw & 7;
    const int tid  = threadIdx.x;
    const int lane = tid & 31;
    const int wave = tid >> 5;
    const int hi   = (lane >> 4) & 1;     // half-wave select
    const int ln   = lane & 15;

    const int    N     = RES_ * RES_;
    const size_t cbase = (size_t)h * HD_;

    // ---- stage K (row-major) and V (transposed) into LDS as bf16 ----
    for (int idx = tid; idx < S_TOK * HD_; idx += 256) {
        int s = idx >> 5, d = idx & 31;
        int y = s / 7, x = s - y * 7;
        size_t g = ((size_t)b * N + (size_t)y * RES_ + wx * WSP_ + x) * CH_ + cbase + d;
        Kb[s * HD_ + d]   = f2bf(k[g]);
        Vt[d * S_PAD + s] = f2bf(v[g]);
    }
    for (int idx = tid + S_TOK * HD_; idx < S_PAD * HD_; idx += 256) {
        int s = idx >> 5, d = idx & 31;
        Kb[s * HD_ + d]   = 0;
        Vt[d * S_PAD + s] = 0;
    }
    __syncthreads();

    // ---- preload LePE depthwise-conv weights for this lane's channels ----
    const int d0 = ln, d1 = ln + 16;
    float wk0[9], wk1[9];
#pragma unroll
    for (int i = 0; i < 9; ++i) {
        wk0[i] = wconv[(size_t)i * CH_ + cbase + d0];
        wk1[i] = wconv[(size_t)i * CH_ + cbase + d1];
    }
    const float bias0 = bconv[cbase + d0];
    const float bias1 = bconv[cbase + d1];

    const float scale = 0.17677669529663687f;   // 32^-0.5
    const int   kb    = hi ? 8 : 0;             // A-frag K base offset

    // load one scaled-Q A-fragment (16 x 32 bf16) for query tile starting at qb
    auto loadQ = [&](int qb) -> FragB16 {
        int sq = qb + ln; if (sq >= S_TOK) sq = S_TOK - 1;   // clamp padded rows
        int yq = sq / 7, xq = sq - yq * 7;
        const float* qrow = q + ((size_t)b * N + (size_t)yq * RES_ + wx * WSP_ + xq) * CH_ + cbase;
        float qv[16];
        *(float4*)&qv[0]  = *(const float4*)(qrow + kb);
        *(float4*)&qv[4]  = *(const float4*)(qrow + kb + 4);
        *(float4*)&qv[8]  = *(const float4*)(qrow + kb + 16);
        *(float4*)&qv[12] = *(const float4*)(qrow + kb + 20);
        FragB16 a;
#pragma unroll
        for (int i = 0; i < 16; ++i) a.us[i] = f2bf(qv[i] * scale);
        return a;
    };

    for (int j = wave; j < NJOBS; j += 8) {
        const int qbase = j * 32;                 // 32 query rows per job

        FragB16 qa0 = loadQ(qbase);
        FragB16 qa1 = loadQ(qbase + 16);

        v8f o00 = {0.f,0.f,0.f,0.f,0.f,0.f,0.f,0.f};
        v8f o01 = o00, o10 = o00, o11 = o00;
        float mrow0[8], lrow0[8], mrow1[8], lrow1[8];
#pragma unroll
        for (int r = 0; r < 8; ++r) {
            mrow0[r] = -1e30f; lrow0[r] = 0.f;
            mrow1[r] = -1e30f; lrow1[r] = 0.f;
        }

        // online-softmax update for one 16-row tile (two 16-col score tiles)
        auto smUpdate = [&](v8f& sA, v8f& sB, float (&mr)[8], float (&lr)[8],
                            v8f& oA, v8f& oB) {
#pragma unroll
            for (int r = 0; r < 8; ++r) {
                float mx    = rowmax16(fmaxf(sA[r], sB[r]));
                float mnew  = fmaxf(mr[r], mx);
                float alpha = __expf(mr[r] - mnew);
                mr[r] = mnew;
                float p0 = __expf(sA[r] - mnew);
                float p1 = __expf(sB[r] - mnew);
                sA[r] = p0; sB[r] = p1;
                float rs = rowsum16(p0 + p1);
                lr[r] = lr[r] * alpha + rs;
                oA[r] *= alpha;
                oB[r] *= alpha;
            }
        };

        auto doChunk = [&](int kc, bool maskTail) {
            const int key0 = kc * 32 + ln;
            const int key1 = key0 + 16;

            // ---- shared B fragments of K^T (reused by both query tiles) ----
            FragB16 bk0, bk1;
            {
                const uint4* p0 = (const uint4*)&Kb[key0 * HD_ + hi * 8];
                bk0.q[0] = p0[0]; bk0.q[1] = p0[2];
                const uint4* p1 = (const uint4*)&Kb[key1 * HD_ + hi * 8];
                bk1.q[0] = p1[0]; bk1.q[1] = p1[2];
            }
            v8f zc = {0.f,0.f,0.f,0.f,0.f,0.f,0.f,0.f};
            v8f s00 = __builtin_amdgcn_wmma_f32_16x16x32_bf16(false, qa0.v, false, bk0.v, (short)0, zc, false, false);
            v8f s01 = __builtin_amdgcn_wmma_f32_16x16x32_bf16(false, qa0.v, false, bk1.v, (short)0, zc, false, false);
            v8f s10 = __builtin_amdgcn_wmma_f32_16x16x32_bf16(false, qa1.v, false, bk0.v, (short)0, zc, false, false);
            v8f s11 = __builtin_amdgcn_wmma_f32_16x16x32_bf16(false, qa1.v, false, bk1.v, (short)0, zc, false, false);

            if (maskTail) {                     // only the last chunk
                if (key0 >= S_TOK) {
#pragma unroll
                    for (int r = 0; r < 8; ++r) { s00[r] = -1e30f; s10[r] = -1e30f; }
                }
                if (key1 >= S_TOK) {
#pragma unroll
                    for (int r = 0; r < 8; ++r) { s01[r] = -1e30f; s11[r] = -1e30f; }
                }
            }

            smUpdate(s00, s01, mrow0, lrow0, o00, o01);
            smUpdate(s10, s11, mrow1, lrow1, o10, o11);

            // ---- D-layout -> A-layout for P via per-wave LDS scratch ----
            unsigned short* pb0 = &Pbuf[wave][0][0];
            unsigned short* pb1 = &Pbuf[wave][1][0];
#pragma unroll
            for (int r = 0; r < 8; ++r) {
                int mr_ = r + hi * 8;
                pb0[mr_ * 32 + ln]      = f2bf(s00[r]);
                pb0[mr_ * 32 + 16 + ln] = f2bf(s01[r]);
                pb1[mr_ * 32 + ln]      = f2bf(s10[r]);
                pb1[mr_ * 32 + 16 + ln] = f2bf(s11[r]);
            }
            FragB16 pa0, pa1;
            {
                const int kb2 = hi ? 8 : 0;
                pa0.q[0] = *(const uint4*)&pb0[ln * 32 + kb2];
                pa0.q[1] = *(const uint4*)&pb0[ln * 32 + kb2 + 16];
                pa1.q[0] = *(const uint4*)&pb1[ln * 32 + kb2];
                pa1.q[1] = *(const uint4*)&pb1[ln * 32 + kb2 + 16];
            }

            // ---- shared B fragments of V (contraction over keys) ----
            FragB16 bv0, bv1;
            {
                const int koff = kc * 32 + hi * 8;
                const uint4* pv0 = (const uint4*)&Vt[d0 * S_PAD + koff];
                bv0.q[0] = pv0[0]; bv0.q[1] = pv0[2];
                const uint4* pv1 = (const uint4*)&Vt[d1 * S_PAD + koff];
                bv1.q[0] = pv1[0]; bv1.q[1] = pv1[2];
            }
            o00 = __builtin_amdgcn_wmma_f32_16x16x32_bf16(false, pa0.v, false, bv0.v, (short)0, o00, false, false);
            o01 = __builtin_amdgcn_wmma_f32_16x16x32_bf16(false, pa0.v, false, bv1.v, (short)0, o01, false, false);
            o10 = __builtin_amdgcn_wmma_f32_16x16x32_bf16(false, pa1.v, false, bv0.v, (short)0, o10, false, false);
            o11 = __builtin_amdgcn_wmma_f32_16x16x32_bf16(false, pa1.v, false, bv1.v, (short)0, o11, false, false);
        };

        for (int kc = 0; kc < KCHUNKS - 1; ++kc) doChunk(kc, false);  // clean chunks
        doChunk(KCHUNKS - 1, true);                                   // masked tail

        // ---- epilogue: normalize, add LePE (3x3 depthwise conv), store ----
        auto epilogue = [&](int qb, v8f& oA, v8f& oB, float (&lr)[8]) {
#pragma unroll
            for (int r = 0; r < 8; ++r) {
                int srow = qb + r + hi * 8;
                if (srow >= S_TOK) continue;
                float invl = 1.0f / lr[r];
                int y = srow / 7, x = srow - y * 7;

                float acc0 = bias0, acc1 = bias1;
#pragma unroll
                for (int ky = 0; ky < 3; ++ky) {
                    int yy = y + ky - 1;
                    if (yy < 0 || yy >= HSP_) continue;
#pragma unroll
                    for (int kx = 0; kx < 3; ++kx) {
                        int xx = x + kx - 1;
                        if (xx < 0 || xx >= WSP_) continue;
                        int sp = yy * 7 + xx;
                        acc0 += bf2f(Vt[d0 * S_PAD + sp]) * wk0[ky * 3 + kx];
                        acc1 += bf2f(Vt[d1 * S_PAD + sp]) * wk1[ky * 3 + kx];
                    }
                }
                size_t g = ((size_t)b * N + (size_t)y * RES_ + wx * WSP_ + x) * CH_ + cbase;
                out[g + d0] = oA[r] * invl + acc0;
                out[g + d1] = oB[r] * invl + acc1;
            }
        };
        epilogue(qbase,      o00, o01, lrow0);
        epilogue(qbase + 16, o10, o11, lrow1);
    }
}

extern "C" void kernel_launch(void* const* d_in, const int* in_sizes, int n_in,
                              void* d_out, int out_size, void* d_ws, size_t ws_size,
                              hipStream_t stream) {
    (void)in_sizes; (void)n_in; (void)out_size; (void)d_ws; (void)ws_size;
    const float* q     = (const float*)d_in[0];
    const float* k     = (const float*)d_in[1];
    const float* v     = (const float*)d_in[2];
    const float* wconv = (const float*)d_in[3];
    const float* bconv = (const float*)d_in[4];
    float* out = (float*)d_out;

    dim3 grid(BATCH_ * NW_, HEADS_);   // 256 windows x 8 heads
    dim3 block(256);                   // 8 waves (wave32)
    lepe_attn_kernel<<<grid, block, 0, stream>>>(q, k, v, wconv, bconv, out);
}
